// vGINEncoderLayerwise_65111704207434
// MI455X (gfx1250) — matrix operand order repaired and checked by reading, not compiled, not run on hardware
//
#include <hip/hip_runtime.h>

#define BN_SCALE 0.9999950000374997f  // 1/sqrt(1+1e-5)

typedef __attribute__((ext_vector_type(16))) __bf16 v16bf;
typedef __attribute__((ext_vector_type(8)))  float  v8f;
typedef __attribute__((ext_vector_type(4)))  unsigned int v4u;
typedef __attribute__((ext_vector_type(8)))  int v8i;
typedef __attribute__((ext_vector_type(4)))  int v4i;

struct __align__(16) B128 { unsigned int x, y, z, w; };
struct __align__(16) F4   { float x, y, z, w; };

union ABu { v16bf v; B128 q[2]; };

__device__ __forceinline__ unsigned short f2bf(float f) {
  union { float f; unsigned int u; } cv; cv.f = f;
  unsigned int r = cv.u + 0x7FFFu + ((cv.u >> 16) & 1u);
  return (unsigned short)(r >> 16);
}

// ---------------------------------------------------------------------------
// TDM: issue one 2-D tile load (rows x 32 bf16) global -> LDS, with the LDS
// pad feature producing a 40-halfword (80 B) row stride to dodge bank
// conflicts.  Tracked on TENSORcnt.  OOB rows (>= rows_valid) read as zero.
// 6-arg builtin flavor (clang-23 / therock-10.0 headers).
// ---------------------------------------------------------------------------
__device__ __forceinline__ void tdm_load_tile(const unsigned short* gsrc,
                                              unsigned int lds_byte_addr,
                                              unsigned int rows_valid,
                                              unsigned int stride_elems) {
  unsigned long long ga = (unsigned long long)(size_t)gsrc;
  v4u g0;
  g0[0] = 1u;                                    // count=1, user descriptor
  g0[1] = lds_byte_addr;                         // LDS dest (bytes)
  g0[2] = (unsigned int)(ga & 0xFFFFFFFFu);      // global_addr[31:0]
  g0[3] = (unsigned int)((ga >> 32) & 0x1FFFFFFu) | (2u << 30);  // [56:32] + type=2

  const unsigned int td0 = 32u;                  // tensor_dim0 (elements/row)
  v8i g1;
  // data_size=1 (2 B), pad_enable, pad_interval=3 (16 DW = 64 B = one row),
  // pad_amount=3 (4 DW = 16 B = 8 halfwords)
  g1[0] = (int)((1u << 16) | (1u << 20) | (3u << 22) | (3u << 25));
  g1[1] = (int)((td0 & 0xFFFFu) << 16);                        // tensor_dim0 lo
  g1[2] = (int)((td0 >> 16) | ((rows_valid & 0xFFFFu) << 16)); // td0 hi | tensor_dim1 lo
  g1[3] = (int)((rows_valid >> 16) | (32u << 16));             // td1 hi | tile_dim0=32
  g1[4] = (int)(128u);                                         // tile_dim1=128, tile_dim2=0
  g1[5] = (int)(stride_elems);                                 // tensor_dim0_stride lo
  g1[6] = 0;                                                   // stride hi, dim1_stride lo
  g1[7] = 0;
  v4i gz4 = {0, 0, 0, 0};                        // 2-D: groups 2/3 unused
  v8i gz8 = {0, 0, 0, 0, 0, 0, 0, 0};            // extra group (6-arg form)
  __builtin_amdgcn_tensor_load_to_lds(g0, g1, gz4, gz4, gz8, 0);
}

// ---------------------------------------------------------------------------
// WMMA GEMM:  out[M,Nout] = epi( A[M,K](bf16) * Bt[Nout,K](bf16)^T )
// epi: y = (acc + bias[n]) * gamma[n]*BN_SCALE + beta[n]; optional ReLU.
// Block tile 128x128, BK=32.  8 waves (4 along M x 2 along N), wave 32x64.
// Double-buffered LDS fed by the Tensor Data Mover (wave 0 issues, TENSORcnt
// FIFO keeps the next tile in flight while the current one is computed).
// ---------------------------------------------------------------------------
__global__ __launch_bounds__(256) void k_gemm_wmma(
    const unsigned short* __restrict__ A,
    const unsigned short* __restrict__ Bt,
    const float* __restrict__ bias,
    const float* __restrict__ gamma,
    const float* __restrict__ beta,
    float* __restrict__ outF,
    unsigned short* __restrict__ outB,
    int Mrows, int K, int Nout, int relu, int out_bf16)
{
  // 2 buffers x (A: 128x40 + B: 128x40) halfwords = 40 KB
  __shared__ unsigned short lds[2 * 2 * 128 * 40];

  const int tid  = threadIdx.x;
  const int lane = tid & 31;
  const int wid  = tid >> 5;
  const int wm   = wid & 3;     // 0..3 : 32-row slice
  const int wn   = wid >> 2;    // 0..1 : 64-col slice
  const int half = lane >> 4;   // 0/1
  const int l15  = lane & 15;
  const int bm   = blockIdx.x;
  const int bn   = blockIdx.y;

  const int nk = K >> 5;
  const int remM = Mrows - bm * 128;
  const unsigned int rowsA = (unsigned int)(remM < 128 ? remM : 128);
  const unsigned short* Abase = A  + (size_t)(bm * 128) * K;
  const unsigned short* Bbase = Bt + (size_t)(bn * 128) * K;
  const unsigned int ldsBase = (unsigned int)(size_t)(&lds[0]);  // flat LDS: addr[31:0]

  v8f acc[2][4];
  const v8f vz = {0.f,0.f,0.f,0.f,0.f,0.f,0.f,0.f};
#pragma unroll
  for (int i = 0; i < 2; ++i)
#pragma unroll
    for (int j = 0; j < 4; ++j) acc[i][j] = vz;

  if (wid == 0) {  // prologue: DMA tile 0 into buffer 0
    tdm_load_tile(Abase, ldsBase, rowsA, (unsigned)K);
    tdm_load_tile(Bbase, ldsBase + 10240u, 128u, (unsigned)K);
  }

  for (int ki = 0; ki < nk; ++ki) {
    const int cur = ki & 1;
    if (wid == 0) {
      if (ki + 1 < nk) {  // prefetch next tile into the other buffer
        const int k0n = (ki + 1) << 5;
        const unsigned int nb = ldsBase + (unsigned)(cur ^ 1) * 20480u;
        tdm_load_tile(Abase + k0n, nb, rowsA, (unsigned)K);
        tdm_load_tile(Bbase + k0n, nb + 10240u, 128u, (unsigned)K);
        __builtin_amdgcn_s_wait_tensorcnt(2);  // current tile done, next in flight
      } else {
        __builtin_amdgcn_s_wait_tensorcnt(0);
      }
    }
    __syncthreads();

    const unsigned short* ldsA = &lds[cur * 10240];
    const unsigned short* ldsB = &lds[cur * 10240 + 5120];

    ABu afr[2], bfr[4];
#pragma unroll
    for (int fm = 0; fm < 2; ++fm) {
      // A 16x32 bf16 layout: lane<16 -> K {0..7,16..23}; lane>=16 -> {8..15,24..31}
      int r = wm * 32 + fm * 16 + l15;
      int base = r * 40 + half * 8;
      afr[fm].q[0] = *(const B128*)(&ldsA[base]);
      afr[fm].q[1] = *(const B128*)(&ldsA[base + 16]);
    }
#pragma unroll
    for (int fn = 0; fn < 4; ++fn) {
      // B 32x16 bf16 layout: lane = column, lane<16 -> K {0..7,16..23}; lane>=16 -> {8..15,24..31}
      int c = wn * 64 + fn * 16 + l15;
      int base = c * 40 + half * 8;
      bfr[fn].q[0] = *(const B128*)(&ldsB[base]);
      bfr[fn].q[1] = *(const B128*)(&ldsB[base + 16]);
    }
#pragma unroll
    for (int fm = 0; fm < 2; ++fm)
#pragma unroll
      for (int fn = 0; fn < 4; ++fn)
        acc[fm][fn] = __builtin_amdgcn_wmma_f32_16x16x32_bf16(
            false, afr[fm].v, false, bfr[fn].v, (short)0, acc[fm][fn], false, false);
    __syncthreads();  // everyone done reading buf[cur] before TDM refills it
  }

  // fused epilogue: bias + inference-BN + optional ReLU, fp32 or bf16 store
#pragma unroll
  for (int fm = 0; fm < 2; ++fm) {
#pragma unroll
    for (int fn = 0; fn < 4; ++fn) {
      int mb = bm * 128 + wm * 32 + fm * 16 + half * 8;
      int nb = bn * 128 + wn * 64 + fn * 16 + l15;
      float g  = gamma[nb] * BN_SCALE;
      float b0 = bias[nb];
      float be = beta[nb];
#pragma unroll
      for (int r = 0; r < 8; ++r) {
        int m = mb + r;
        if (m < Mrows) {
          float y = (acc[fm][fn][r] + b0) * g + be;
          if (relu) y = fmaxf(y, 0.f);
          if (out_bf16) outB[(size_t)m * Nout + nb] = f2bf(y);
          else          outF[(size_t)m * Nout + nb] = y;
        }
      }
    }
  }
}

// ---------------------------------------------------------------------------
// support kernels
// ---------------------------------------------------------------------------
__global__ void k_zero_f32(float* __restrict__ p, long long n) {
  long long i = (long long)blockIdx.x * 256 + threadIdx.x;
  if (i < n) p[i] = 0.f;
}

// W[K,Ncols] fp32 -> Wt[Ncols,K] bf16 (so WMMA B loads contiguous K per column)
__global__ void k_transpose_cvt(const float* __restrict__ W,
                                unsigned short* __restrict__ Wt, int K, int Ncols) {
  long long idx = (long long)blockIdx.x * 256 + threadIdx.x;
  if (idx >= (long long)K * Ncols) return;
  int n = (int)(idx / K);
  int k = (int)(idx - (long long)n * K);
  Wt[idx] = f2bf(W[(size_t)k * Ncols + n]);
}

__global__ void k_scatter_add(const float* __restrict__ h, const int* __restrict__ src,
                              const int* __restrict__ dst, float* __restrict__ agg,
                              int F, long long total) {
  long long idx = (long long)blockIdx.x * 256 + threadIdx.x;
  if (idx >= total) return;
  int cpv = F >> 2;
  int e = (int)(idx / cpv);
  int c = (int)(idx - (long long)e * cpv) << 2;
  int s = src[e], d = dst[e];
  F4 v = *(const F4*)(h + (size_t)s * F + c);
  float* p = agg + (size_t)d * F + c;
  atomicAdd(p + 0, v.x); atomicAdd(p + 1, v.y);
  atomicAdd(p + 2, v.z); atomicAdd(p + 3, v.w);
}

__global__ void k_add_cvt(const float* __restrict__ a, const float* __restrict__ b,
                          unsigned short* __restrict__ o, long long n) {
  long long i = (long long)blockIdx.x * 256 + threadIdx.x;
  if (i < n) o[i] = f2bf(a[i] + b[i]);
}

__global__ void k_add_vfeat(const float* __restrict__ h, const float* __restrict__ vf,
                            const int* __restrict__ batch, float* __restrict__ o,
                            int F, long long n) {
  long long i = (long long)blockIdx.x * 256 + threadIdx.x;
  if (i >= n) return;
  int node = (int)(i / F);
  int f = (int)(i - (long long)node * F);
  o[i] = h[i] + vf[(size_t)batch[node] * F + f];
}

__global__ void k_broadcast(const float* __restrict__ vemb, float* __restrict__ vf,
                            int n, int Hm) {
  int i = blockIdx.x * 256 + threadIdx.x;
  if (i < n) vf[i] = vemb[i % Hm];
}

__global__ void k_pool(const float* __restrict__ h, const int* __restrict__ batch,
                       float* __restrict__ pooled, int F, long long total) {
  long long idx = (long long)blockIdx.x * 256 + threadIdx.x;
  if (idx >= total) return;
  int cpv = F >> 2;
  int node = (int)(idx / cpv);
  int c = (int)(idx - (long long)node * cpv) << 2;
  int g = batch[node];
  F4 v = *(const F4*)(h + (size_t)node * F + c);
  float* p = pooled + (size_t)g * F + c;
  atomicAdd(p + 0, v.x); atomicAdd(p + 1, v.y);
  atomicAdd(p + 2, v.z); atomicAdd(p + 3, v.w);
}

__global__ void k_counts(const int* __restrict__ batch, float* __restrict__ cnt, int n) {
  int i = blockIdx.x * 256 + threadIdx.x;
  if (i < n) atomicAdd(&cnt[batch[i]], 1.f);
}

__global__ void k_finalize(const float* __restrict__ sums, const float* __restrict__ cnt,
                           float* __restrict__ out, int Hm, int n) {
  int i = blockIdx.x * 256 + threadIdx.x;
  if (i < n) out[i] = sums[i] / fmaxf(cnt[i / Hm], 1.f);
}

// ---------------------------------------------------------------------------
extern "C" void kernel_launch(void* const* d_in, const int* in_sizes, int n_in,
                              void* d_out, int out_size, void* d_ws, size_t ws_size,
                              hipStream_t stream) {
  const int N = 50000, E = 800000, G = 128, D = 128, H = 512, H2 = 1024, L = 4;
  const float* x      = (const float*)d_in[0];
  const int*   src    = (const int*)d_in[1];
  const int*   dstp   = src + E;
  const int*   batch  = (const int*)d_in[2];
  const float* c1_W1  = (const float*)d_in[3];
  const float* c1_b1  = (const float*)d_in[4];
  const float* c1_g1  = (const float*)d_in[5];
  const float* c1_be1 = (const float*)d_in[6];
  const float* c1_W2  = (const float*)d_in[7];
  const float* c1_b2  = (const float*)d_in[8];
  const float* bn1_g  = (const float*)d_in[9];
  const float* bn1_b  = (const float*)d_in[10];
  const float* cs_W1  = (const float*)d_in[11];
  const float* cs_b1  = (const float*)d_in[12];
  const float* cs_g1  = (const float*)d_in[13];
  const float* cs_be1 = (const float*)d_in[14];
  const float* cs_W2  = (const float*)d_in[15];
  const float* cs_b2  = (const float*)d_in[16];
  const float* bns_g  = (const float*)d_in[17];
  const float* bns_b  = (const float*)d_in[18];
  const float* vemb   = (const float*)d_in[19];
  const float* v_W1   = (const float*)d_in[20];
  const float* v_b1   = (const float*)d_in[21];
  const float* v_g1   = (const float*)d_in[22];
  const float* v_be1  = (const float*)d_in[23];
  const float* v_W2   = (const float*)d_in[24];
  const float* v_b2   = (const float*)d_in[25];
  const float* v_g2   = (const float*)d_in[26];
  const float* v_be2  = (const float*)d_in[27];
  float* out = (float*)d_out;
  (void)in_sizes; (void)n_in; (void)out_size; (void)ws_size;

  size_t off = 0;
  auto alloc = [&](size_t bytes) -> void* {
    void* p = (char*)d_ws + off;
    off += (bytes + 255) & ~(size_t)255;
    return p;
  };
  unsigned short* c1_W1t = (unsigned short*)alloc((size_t)D * H2 * 2);
  unsigned short* c1_W2t = (unsigned short*)alloc((size_t)H2 * H * 2);
  unsigned short* cs_W1t = (unsigned short*)alloc((size_t)L * H * H2 * 2);
  unsigned short* cs_W2t = (unsigned short*)alloc((size_t)L * H2 * H * 2);
  unsigned short* v_W1t  = (unsigned short*)alloc((size_t)H * H2 * 2);
  unsigned short* v_W2t  = (unsigned short*)alloc((size_t)H2 * H * 2);
  float* vfeat  = (float*)alloc((size_t)G * H * 4);
  float* pooled = (float*)alloc((size_t)G * H * 4);
  unsigned short* zbv = (unsigned short*)alloc((size_t)G * H * 2);
  unsigned short* tv  = (unsigned short*)alloc((size_t)G * H2 * 2);
  float* cnt = (float*)alloc((size_t)G * 4);
  float* hin = (float*)alloc((size_t)N * H * 4);
  float* agg = (float*)alloc((size_t)N * H * 4);
  unsigned short* zb = (unsigned short*)alloc((size_t)N * H * 2);
  unsigned short* t1 = (unsigned short*)alloc((size_t)N * H2 * 2);
  float* hbuf = (float*)alloc((size_t)N * H * 4);

  auto eb = [](long long n) { return dim3((unsigned)((n + 255) / 256)); };
  const int MB = (N + 127) / 128;

  // ---- weights: transpose + fp32->bf16 ----
  k_transpose_cvt<<<eb((long long)D * H2), 256, 0, stream>>>(c1_W1, c1_W1t, D, H2);
  k_transpose_cvt<<<eb((long long)H2 * H), 256, 0, stream>>>(c1_W2, c1_W2t, H2, H);
  for (int i = 0; i < L; ++i) {
    k_transpose_cvt<<<eb((long long)H * H2), 256, 0, stream>>>(
        cs_W1 + (size_t)i * H * H2, cs_W1t + (size_t)i * H2 * H, H, H2);
    k_transpose_cvt<<<eb((long long)H2 * H), 256, 0, stream>>>(
        cs_W2 + (size_t)i * H2 * H, cs_W2t + (size_t)i * H * H2, H2, H);
  }
  k_transpose_cvt<<<eb((long long)H * H2), 256, 0, stream>>>(v_W1, v_W1t, H, H2);
  k_transpose_cvt<<<eb((long long)H2 * H), 256, 0, stream>>>(v_W2, v_W2t, H2, H);

  // ---- virtual-node features: broadcast embedding row 0 ----
  k_broadcast<<<eb((long long)G * H), 256, 0, stream>>>(vemb, vfeat, G * H, H);

  // ---- layer 1: conv1 (D -> 2H -> H) + BN1 + ReLU ----
  k_zero_f32<<<eb((long long)N * D), 256, 0, stream>>>(agg, (long long)N * D);
  k_scatter_add<<<eb((long long)E * (D / 4)), 256, 0, stream>>>(
      x, src, dstp, agg, D, (long long)E * (D / 4));
  k_add_cvt<<<eb((long long)N * D), 256, 0, stream>>>(x, agg, zb, (long long)N * D);
  k_gemm_wmma<<<dim3(MB, H2 / 128), 256, 0, stream>>>(
      zb, c1_W1t, c1_b1, c1_g1, c1_be1, nullptr, t1, N, D, H2, 1, 1);
  k_gemm_wmma<<<dim3(MB, H / 128), 256, 0, stream>>>(
      t1, c1_W2t, c1_b2, bn1_g, bn1_b, hbuf, nullptr, N, H2, H, 1, 0);

  // ---- L extra GIN layers with virtual node ----
  for (int i = 0; i < L; ++i) {
    k_add_vfeat<<<eb((long long)N * H), 256, 0, stream>>>(
        hbuf, vfeat, batch, hin, H, (long long)N * H);
    k_zero_f32<<<eb((long long)N * H), 256, 0, stream>>>(agg, (long long)N * H);
    k_scatter_add<<<eb((long long)E * (H / 4)), 256, 0, stream>>>(
        hin, src, dstp, agg, H, (long long)E * (H / 4));
    k_add_cvt<<<eb((long long)N * H), 256, 0, stream>>>(hin, agg, zb, (long long)N * H);
    k_gemm_wmma<<<dim3(MB, H2 / 128), 256, 0, stream>>>(
        zb, cs_W1t + (size_t)i * H2 * H, cs_b1 + i * H2, cs_g1 + i * H2,
        cs_be1 + i * H2, nullptr, t1, N, H, H2, 1, 1);
    k_gemm_wmma<<<dim3(MB, H / 128), 256, 0, stream>>>(
        t1, cs_W2t + (size_t)i * H * H2, cs_b2 + i * H, bns_g + i * H,
        bns_b + i * H, hbuf, nullptr, N, H2, H, (i < L - 1) ? 1 : 0, 0);
    if (i < L - 1) {
      k_zero_f32<<<eb((long long)G * H), 256, 0, stream>>>(pooled, (long long)G * H);
      k_pool<<<eb((long long)N * (H / 4)), 256, 0, stream>>>(
          hbuf, batch, pooled, H, (long long)N * (H / 4));
      k_add_cvt<<<eb((long long)G * H), 256, 0, stream>>>(
          pooled, vfeat, zbv, (long long)G * H);
      k_gemm_wmma<<<dim3(1, H2 / 128), 256, 0, stream>>>(
          zbv, v_W1t, v_b1, v_g1, v_be1, nullptr, tv, G, H, H2, 1, 1);
      k_gemm_wmma<<<dim3(1, H / 128), 256, 0, stream>>>(
          tv, v_W2t, v_b2, v_g2, v_be2, vfeat, nullptr, G, H2, H, 1, 0);
    }
  }

  // ---- global mean pool readout ----
  k_zero_f32<<<eb((long long)G * H), 256, 0, stream>>>(pooled, (long long)G * H);
  k_zero_f32<<<eb((long long)G), 256, 0, stream>>>(cnt, (long long)G);
  k_pool<<<eb((long long)N * (H / 4)), 256, 0, stream>>>(
      hbuf, batch, pooled, H, (long long)N * (H / 4));
  k_counts<<<eb((long long)N), 256, 0, stream>>>(batch, cnt, N);
  k_finalize<<<eb((long long)G * H), 256, 0, stream>>>(pooled, cnt, out, H, G * H);
}
